// ConvolutionalKAN_19945828123052
// MI455X (gfx1250) — compile-verified, matrix operand order checked by reading, not run on hardware
//
#include <hip/hip_runtime.h>
#include <hip/hip_bf16.h>

typedef __attribute__((ext_vector_type(16))) _Float16 v16h;
typedef __attribute__((ext_vector_type(8)))  float    v8f;
typedef int v4i_vs __attribute__((vector_size(16)));   // matches builtin proto

#define AS1 __attribute__((address_space(1)))
#define AS3 __attribute__((address_space(3)))

// ---------------------------------------------------------------------------
// Problem constants (from reference): x[16,64,64,64], coeff[64,64,3,3,8]
// ---------------------------------------------------------------------------
#define BATCH   16
#define CIN     64
#define COUT    64
#define HW      64
#define NB      8
#define KS      3
#define HOUT    62
#define WOUT    62
#define NPIX    (HOUT * WOUT)          // 3844 output pixels per batch image
#define KC      (CIN * NB)             // 512 "channels" per tap
#define NCHUNK  72                     // 9 taps * (512/64) K-chunks

// phi layout (f16): [b][h][w][Cin*8]   -> 64 MB (fits in 192 MB L2)
// wt  layout (f16): [tap(9)][Cout][Cin*8]

// ---------------------------------------------------------------------------
// CDNA5 async global->LDS copy (16 B per lane), tracked by ASYNCcnt.
// IOFFSET is added to BOTH the global and LDS address (ISA 10.x / 15.18.3).
// Builtin proto (probe-confirmed): (v4i AS1*, v4i AS3*, Imm offset, Imm cpol).
// Host pass only parses device bodies -> give it a no-op stub.
// ---------------------------------------------------------------------------
#if defined(__HIP_DEVICE_COMPILE__)
#  if __has_builtin(__builtin_amdgcn_global_load_async_to_lds_b128)
#    define ASYNC_CP16(lds, g, OFF)                                           \
        __builtin_amdgcn_global_load_async_to_lds_b128(                       \
            (AS1 v4i_vs*)(g), (AS3 v4i_vs*)(lds), (OFF), 0)
#  else
#    define ASYNC_CP16(lds, g, OFF)                                           \
        asm volatile("global_load_async_to_lds_b128 %0, %1, off offset:%2"    \
                     :: "v"((unsigned)(size_t)(AS3 char*)(void*)(lds)),       \
                        "v"((const void*)(g)), "n"(OFF) : "memory")
#  endif
#else
#  define ASYNC_CP16(lds, g, OFF) ((void)(lds), (void)(g))
#endif

__device__ __forceinline__ void wait_async_le6() {
#if defined(__HIP_DEVICE_COMPILE__)
#  if __has_builtin(__builtin_amdgcn_s_wait_asynccnt)
    __builtin_amdgcn_s_wait_asynccnt(6);
#  else
    asm volatile("s_wait_asynccnt 0x6" ::: "memory");
#  endif
#endif
}

// ---------------------------------------------------------------------------
// Pass 1: cubic B-spline basis (Cox-de Boor, uniform knots i/11), x -> phi f16
// ---------------------------------------------------------------------------
__device__ __forceinline__ void bspline8(float x, float* o) {
    const float s = 1.0f / 11.0f;               // knot spacing
    float B[11];
#pragma unroll
    for (int i = 0; i < 11; ++i) {
        float k0 = (float)i * s, k1 = (float)(i + 1) * s;
        B[i] = (k0 <= x && x < k1) ? 1.0f : 0.0f;
    }
#pragma unroll
    for (int d = 1; d <= 3; ++d) {
        float invd = 11.0f / (float)d;          // 1/(k[i+d]-k[i]) uniform
#pragma unroll
        for (int i = 0; i + d < 11; ++i) {      // ascending: reads old B[i+1]
            float ki   = (float)i * s;
            float kid1 = (float)(i + d + 1) * s;
            B[i] = (x - ki) * invd * B[i] + (kid1 - x) * invd * B[i + 1];
        }
    }
#pragma unroll
    for (int i = 0; i < 8; ++i) o[i] = B[i];
}

__global__ void kan_phi_kernel(const float* __restrict__ x,
                               _Float16* __restrict__ phi) {
    int idx = blockIdx.x * blockDim.x + threadIdx.x;   // over [b][j][h][w]
    int w = idx & 63;
    int h = (idx >> 6) & 63;
    int j = (idx >> 12) & 63;
    int b = idx >> 18;
    float bs[8];
    bspline8(x[idx], bs);
    union { uint4 q; _Float16 h[8]; } u;
#pragma unroll
    for (int m = 0; m < 8; ++m) u.h[m] = (_Float16)bs[m];
    size_t off = ((((size_t)b * HW + h) * HW + w) * KC) + (size_t)j * NB;
    *(uint4*)(phi + off) = u.q;                        // 16B contiguous store
}

// ---------------------------------------------------------------------------
// Pass 2: repack coeff [o][j][k][l][m] f32 -> wt [k*3+l][o][j*8+m] f16
// (B-matrix N-major layout so WMMA B fragments read contiguously in K)
// ---------------------------------------------------------------------------
__global__ void kan_wt_kernel(const float* __restrict__ coeff,
                              _Float16* __restrict__ wt) {
    int i = blockIdx.x * blockDim.x + threadIdx.x;     // flat coeff index
    if (i >= COUT * CIN * KS * KS * NB) return;
    int m = i & 7;
    int l = (i >> 3) % 3;
    int k = (i / 24) % 3;
    int j = (i / 72) & 63;
    int o = i / 4608;
    int kk = k * 3 + l;
    wt[((size_t)(kk * COUT + o) << 9) + j * NB + m] = (_Float16)coeff[i];
}

// ---------------------------------------------------------------------------
// Pass 3: implicit-GEMM KAN conv via v_wmma_f32_16x16x32_f16.
//   Block: 256 threads (8 waves). M-tile = 128 output pixels, N = 64 (Cout).
//   Double-buffered LDS fed by GLOBAL_LOAD_ASYNC_TO_LDS_B128 (ASYNCcnt).
// ---------------------------------------------------------------------------
#define MTILE   128
#define APITCH  72            // 64 + 8 halves pad (144 B) breaks bank conflicts

__global__ __launch_bounds__(256, 2)
void kan_gemm_kernel(const _Float16* __restrict__ phi,
                     const _Float16* __restrict__ wt,
                     float* __restrict__ out) {
    __shared__ __align__(16) _Float16 As[2][MTILE * APITCH];  // 2 x 18,432 B
    __shared__ __align__(16) _Float16 Bs[2][COUT  * APITCH];  // 2 x  9,216 B

    const int b    = blockIdx.y;
    const int m0   = blockIdx.x * MTILE;
    const int tid  = threadIdx.x;
    const int lane = tid & 31;
    const int wave = tid >> 5;
    const int mrow = lane & 15;           // M (A) / N (B) index within tile
    const int msel = lane >> 4;           // half-wave K-select per ISA layout

    const v8f vzero = {0.f,0.f,0.f,0.f,0.f,0.f,0.f,0.f};
    v8f acc[4] = {vzero, vzero, vzero, vzero};

    const size_t phi_b = (size_t)b * HW * HW * KC;

    // A staging: thread covers 64 B of one row (row = tid/2, 4 async b128).
    const int ar = tid >> 1;
    const int aq = (tid & 1) * 32;                 // starting half in row
    const int ap = m0 + ar;
    const int apc = ap > (NPIX - 1) ? (NPIX - 1) : ap;   // clamp, no divergence
    const int ah = apc / WOUT, aw = apc % WOUT;
    // B staging: thread covers 32 B of one row (row = tid/4, 2 async b128).
    const int br = tid >> 2;
    const int bq = (tid & 3) * 16;                 // starting half in row

    // Issue the 6 async copies for chunk t into LDS buffer `buf`.
    auto stage = [&](int t, int buf) {
        const int kk  = t >> 3;                    // tap 0..8
        const int cof = (t & 7) << 6;              // channel offset 0..448
        const _Float16* ag = phi + phi_b +
            (((size_t)(ah + kk / 3) * HW) + (aw + kk % 3)) * KC + cof + aq;
        _Float16* al = &As[buf][ar * APITCH + aq];
        ASYNC_CP16(al, ag, 0);
        ASYNC_CP16(al, ag, 16);
        ASYNC_CP16(al, ag, 32);
        ASYNC_CP16(al, ag, 48);
        const _Float16* bg = wt + ((size_t)(kk * COUT + br) << 9) + cof + bq;
        _Float16* bl = &Bs[buf][br * APITCH + bq];
        ASYNC_CP16(bl, bg, 0);
        ASYNC_CP16(bl, bg, 16);
    };

    stage(0, 0);                                   // prologue: 6 ops in flight

    for (int t = 0; t < NCHUNK; ++t) {
        // Issue next chunk into the other buffer (safe: barrier at end of the
        // previous iteration guaranteed all waves finished reading it).
        // At t == NCHUNK-1 re-issue chunk 71 redundantly to keep the
        // in-flight count uniform (written to the dead buffer, never read).
        stage(t + 1 < NCHUNK ? t + 1 : NCHUNK - 1, (t + 1) & 1);

        // ASYNCcnt completes in order: <=6 left  =>  chunk t's 6 ops landed.
        wait_async_le6();
        __syncthreads();

        // ---- 2 K-steps of 32, 4 N-tiles of 16 -> 8 WMMAs per chunk per wave
        const _Float16* arow = &As[t & 1][(wave * 16 + mrow) * APITCH];
        const _Float16* bbase = &Bs[t & 1][0];
#pragma unroll
        for (int ks = 0; ks < 2; ++ks) {
            const int kbase = ks * 32 + msel * 8;  // ISA 16-bit A/B lane layout
            union { uint4 q[2]; v16h v; } ua;
            ua.q[0] = *(const uint4*)(arow + kbase);
            ua.q[1] = *(const uint4*)(arow + kbase + 16);
#pragma unroll
            for (int nt = 0; nt < 4; ++nt) {
                union { uint4 q[2]; v16h v; } ub;
                const _Float16* brow = bbase + (nt * 16 + mrow) * APITCH;
                ub.q[0] = *(const uint4*)(brow + kbase);
                ub.q[1] = *(const uint4*)(brow + kbase + 16);
                acc[nt] = __builtin_amdgcn_wmma_f32_16x16x32_f16(
                    false, ua.v, false, ub.v, (short)0, acc[nt], false, false);
            }
        }
        __syncthreads();
    }

    // ---- epilogue: C/D layout -> lane m = msel*8 + v, n = mrow
    const int prow = m0 + wave * 16 + msel * 8;
#pragma unroll
    for (int nt = 0; nt < 4; ++nt) {
        const int o = nt * 16 + mrow;
        float* outb = out + ((size_t)b * COUT + o) * NPIX;
#pragma unroll
        for (int v = 0; v < 8; ++v) {
            int p = prow + v;
            if (p < NPIX) outb[p] = acc[nt][v];
        }
    }
}

// ---------------------------------------------------------------------------
extern "C" void kernel_launch(void* const* d_in, const int* in_sizes, int n_in,
                              void* d_out, int out_size, void* d_ws, size_t ws_size,
                              hipStream_t stream) {
    (void)in_sizes; (void)n_in; (void)out_size; (void)ws_size;
    const float* x     = (const float*)d_in[0];
    const float* coeff = (const float*)d_in[1];
    float*       out   = (float*)d_out;

    _Float16* phi = (_Float16*)d_ws;                                // 64 MB
    _Float16* wtb = (_Float16*)d_ws + (size_t)BATCH * HW * HW * KC; // 0.6 MB

    // Pass 1: basis expansion (exact 16384 blocks, no tail)
    kan_phi_kernel<<<(BATCH * CIN * HW * HW) / 256, 256, 0, stream>>>(x, phi);
    // Pass 2: weight repack
    kan_wt_kernel<<<(COUT * CIN * KS * KS * NB + 255) / 256, 256, 0, stream>>>(coeff, wtb);
    // Pass 3: WMMA implicit GEMM, grid = (ceil(3844/128)=31 M-tiles, 16 batches)
    dim3 grid((NPIX + MTILE - 1) / MTILE, BATCH);
    kan_gemm_kernel<<<grid, 256, 0, stream>>>(phi, wtb, out);
}
